// GraphConvBlock_51427938402642
// MI455X (gfx1250) — compile-verified, matrix operand order checked by reading, not run on hardware
//
#include <hip/hip_runtime.h>

typedef __attribute__((ext_vector_type(16))) _Float16 v16h;
typedef __attribute__((ext_vector_type(8)))  _Float16 v8h;
typedef __attribute__((ext_vector_type(4)))  _Float16 v4h;
typedef __attribute__((ext_vector_type(8)))  float    v8f;

#define C_IN   64
#define C_OUTC 64
#define KP     16
#define KC     (KP * C_IN)   // 1024 = flattened (kernel, c_in) contraction dim
#define QT     16            // queries per workgroup (= WMMA M)
#define BN_EPS 1e-5f

__device__ __forceinline__ int lower_bound_i32(const int* __restrict__ a, int n, int key) {
  int lo = 0, hi = n;
  while (lo < hi) { int mid = (lo + hi) >> 1; if (a[mid] < key) lo = mid + 1; else hi = mid; }
  return lo;
}

// Pass 0: permute kernel weights [1024][64] f32 into f16 *fragment-major* order:
//   bfrag[((kk*2 + kb)*64 + ncol)*16 + e] = W[kk*32 + kb*8 + (e<8 ? e : e+8)][ncol]
// so each WMMA B fragment is one contiguous 32-byte run per lane.
// Also zeroes the 128-float BN-stats accumulator.
__global__ void prep_kernel(const float* __restrict__ w, _Float16* __restrict__ bfrag,
                            float* __restrict__ stats, int nw) {
  int i = blockIdx.x * blockDim.x + threadIdx.x;
  if (i < nw) {
    const int e    = i & 15;
    const int ncol = (i >> 4) & 63;
    const int kb   = (i >> 10) & 1;
    const int kk   = i >> 11;
    const int k    = kk * 32 + kb * 8 + (e < 8 ? e : e + 8);
    bfrag[i] = (_Float16)w[k * C_OUTC + ncol];
  }
  if (i < 2 * C_OUTC) stats[i] = 0.0f;
}

// Pass 1 (fused): per-block = 16 consecutive queries.
//  - binary search edge range in sorted e_query
//  - edge loop: top-3 kernel points + softmax, ds_add_f32 scatter into LDS s_tile[16][1024]
//  - one-shot f32->f16 conversion of the tile (packed cvt, LDS->LDS)
//  - 4 waves x v_wmma_f32_16x16x32_f16 over K=1024; A via 2x ds_load_b128,
//    B via 2x global_load_b128 from the fragment-major weight buffer
//  - divide by per-query edge count (boundary shuffles + v_rcp), BN partial sums
__global__ __launch_bounds__(128) void fused_conv_kernel(
    const float* __restrict__ ref_bxyz, const float* __restrict__ ref_feat,
    const float* __restrict__ query_bxyz, const int* __restrict__ e_ref,
    const int* __restrict__ e_query, const float* __restrict__ kernel_pos,
    const _Float16* __restrict__ bfrag, float* __restrict__ out,
    float* __restrict__ stats, int M, int E)
{
  __shared__ __align__(16) float    s_tile[QT * KC];  // 64 KB f32 accumulator
  __shared__ __align__(16) _Float16 s_h[QT * KC];     // 32 KB f16 staging
  const int tid  = threadIdx.x;
  const int lane = tid & 31;
  const int wave = tid >> 5;
  const int q0   = blockIdx.x * QT;

  // Lane l (l<=16) holds lower_bound(q0 + l); counts derive from adjacent diffs.
  const int lbv = lower_bound_i32(e_query, E, q0 + (lane < 16 ? lane : 16));
  const int lo  = __shfl(lbv, 0);
  const int hi  = __shfl(lbv, 16);

  // zero LDS accumulator
  float4* s4 = (float4*)s_tile;
  for (int i = tid; i < QT * KC / 4; i += 128)
    s4[i] = make_float4(0.f, 0.f, 0.f, 0.f);
  __syncthreads();

  // kernel point positions in registers (48 floats, uniform)
  float kpx[KP], kpy[KP], kpz[KP];
#pragma unroll
  for (int k = 0; k < KP; ++k) {
    kpx[k] = kernel_pos[3 * k + 0];
    kpy[k] = kernel_pos[3 * k + 1];
    kpz[k] = kernel_pos[3 * k + 2];
  }

  const float4* feat4 = (const float4*)ref_feat;
  for (int e = lo + tid; e < hi; e += 128) {
    const int q  = e_query[e];
    const int r  = e_ref[e];
    const int lq = q - q0;
    const float dx = ref_bxyz[4 * r + 1] - query_bxyz[4 * q + 1];
    const float dy = ref_bxyz[4 * r + 2] - query_bxyz[4 * q + 2];
    const float dz = ref_bxyz[4 * r + 3] - query_bxyz[4 * q + 3];

    // top-3 smallest squared distances
    float d0 = 3.4e38f, d1 = 3.4e38f, d2 = 3.4e38f;
    int   k0 = 0, k1 = 0, k2 = 0;
#pragma unroll
    for (int k = 0; k < KP; ++k) {
      const float ax = dx - kpx[k], ay = dy - kpy[k], az = dz - kpz[k];
      const float d  = ax * ax + ay * ay + az * az;
      if (d < d0)      { d2 = d1; k2 = k1; d1 = d0; k1 = k0; d0 = d; k0 = k; }
      else if (d < d1) { d2 = d1; k2 = k1; d1 = d;  k1 = k; }
      else if (d < d2) { d2 = d;  k2 = k; }
    }
    // softmax over -d2 (max-subtracted)
    float w1 = __expf(d0 - d1);
    float w2 = __expf(d0 - d2);
    const float inv = 1.0f / (1.0f + w1 + w2);
    const float w0  = inv; w1 *= inv; w2 *= inv;

    float* srow = s_tile + lq * KC;
    const int b0 = k0 * C_IN, b1 = k1 * C_IN, b2 = k2 * C_IN;
#pragma unroll 4
    for (int c4 = 0; c4 < 16; ++c4) {
      const float4 f = feat4[r * 16 + c4];
      const int o = c4 * 4;
      atomicAdd(&srow[b0 + o + 0], w0 * f.x); atomicAdd(&srow[b0 + o + 1], w0 * f.y);
      atomicAdd(&srow[b0 + o + 2], w0 * f.z); atomicAdd(&srow[b0 + o + 3], w0 * f.w);
      atomicAdd(&srow[b1 + o + 0], w1 * f.x); atomicAdd(&srow[b1 + o + 1], w1 * f.y);
      atomicAdd(&srow[b1 + o + 2], w1 * f.z); atomicAdd(&srow[b1 + o + 3], w1 * f.w);
      atomicAdd(&srow[b2 + o + 0], w2 * f.x); atomicAdd(&srow[b2 + o + 1], w2 * f.y);
      atomicAdd(&srow[b2 + o + 2], w2 * f.z); atomicAdd(&srow[b2 + o + 3], w2 * f.w);
    }
  }
  __syncthreads();

  // one-shot f32 -> f16 conversion of the accumulated tile (packed converts)
  for (int i = tid; i < QT * KC / 4; i += 128) {
    const float4 f = s4[i];
    v4h h; h[0] = (_Float16)f.x; h[1] = (_Float16)f.y;
           h[2] = (_Float16)f.z; h[3] = (_Float16)f.w;
    *(v4h*)(s_h + 4 * i) = h;
  }
  __syncthreads();

  // ---- WMMA GEMM: [16 x 1024] (LDS f16) x [1024 x 64] (fragment-major f16) ----
  const int kb   = (lane & 16) ? 1 : 0;     // lane-half -> K sub-block select
  const int mrow = lane & 15;               // A row (local query)
  const int ncol = wave * 16 + (lane & 15); // B column / D column

  const _Float16* __restrict__ s_row = s_h + mrow * KC + kb * 8;
  const _Float16* __restrict__ b_ptr = bfrag + ((size_t)(kb * 64 + ncol) << 4);

  v8f acc = {0.f, 0.f, 0.f, 0.f, 0.f, 0.f, 0.f, 0.f};
#pragma unroll 4
  for (int kk = 0; kk < KC / 32; ++kk) {
    const v8h a_lo = *(const v8h*)(s_row + kk * 32);
    const v8h a_hi = *(const v8h*)(s_row + kk * 32 + 16);
    const v16h av  = __builtin_shufflevector(a_lo, a_hi,
        0, 1, 2, 3, 4, 5, 6, 7, 8, 9, 10, 11, 12, 13, 14, 15);
    const v16h bv  = *(const v16h*)(b_ptr + (size_t)kk * 2048);
    acc = __builtin_amdgcn_wmma_f32_16x16x32_f16(
        /*neg_a=*/false, av, /*neg_b=*/false, bv,
        /*c_mod=*/(short)0, acc, /*reuse_a=*/false, /*reuse_b=*/false);
  }

  // ---- scatter-mean divide (v_rcp), store, BN partial statistics ----
  float psum = 0.f, psq = 0.f;
#pragma unroll
  for (int j = 0; j < 8; ++j) {
    const int m  = j + ((lane & 16) ? 8 : 0);         // D-row -> local query
    const int c0 = __shfl(lbv, m);
    const int c1 = __shfl(lbv, m + 1);
    const float rden = __builtin_amdgcn_rcpf(fmaxf((float)(c1 - c0), 1.0f));
    const float val  = acc[j] * rden;
    const int q = q0 + m;
    if (q < M) out[q * C_OUTC + ncol] = val;
    psum += val;
    psq  += val * val;
  }
  // lanes l and l^16 share the same channel -> combine, then 16 atomics per wave
  psum += __shfl_xor(psum, 16);
  psq  += __shfl_xor(psq, 16);
  if (lane < 16) {
    atomicAdd(&stats[ncol], psum);
    atomicAdd(&stats[C_OUTC + ncol], psq);
  }
}

// Pass 2: BatchNorm (training-mode batch stats) in place.
__global__ void bn_kernel(float* __restrict__ out, const float* __restrict__ stats,
                          const float* __restrict__ gamma, const float* __restrict__ beta,
                          int M) {
  const int i = blockIdx.x * blockDim.x + threadIdx.x;
  if (i >= M * C_OUTC) return;
  const int n = i & (C_OUTC - 1);
  const float invM = 1.0f / (float)M;
  const float mean = stats[n] * invM;
  const float var  = stats[C_OUTC + n] * invM - mean * mean;
  const float sc   = rsqrtf(var + BN_EPS) * gamma[n];
  out[i] = (out[i] - mean) * sc + beta[n];
}

extern "C" void kernel_launch(void* const* d_in, const int* in_sizes, int n_in,
                              void* d_out, int out_size, void* d_ws, size_t ws_size,
                              hipStream_t stream) {
  const float* ref_bxyz       = (const float*)d_in[0];
  const float* ref_feat       = (const float*)d_in[1];
  const float* query_bxyz     = (const float*)d_in[2];
  const int*   e_ref          = (const int*)d_in[3];
  const int*   e_query        = (const int*)d_in[4];
  const float* kernel_pos     = (const float*)d_in[5];
  const float* kernel_weights = (const float*)d_in[6];
  const float* bn_gamma       = (const float*)d_in[7];
  const float* bn_beta        = (const float*)d_in[8];
  (void)n_in; (void)out_size; (void)ws_size;

  const int M = in_sizes[2] / 4;
  const int E = in_sizes[3];

  float*     out   = (float*)d_out;
  _Float16*  bfrag = (_Float16*)d_ws;                                    // 128 KB
  float*     stats = (float*)((char*)d_ws + (size_t)KC * C_OUTC * 2);    // 512 B

  const int nw = KC * C_OUTC;  // 65536 weight elements
  prep_kernel<<<(nw + 255) / 256, 256, 0, stream>>>(kernel_weights, bfrag, stats, nw);

  const int nblocks = (M + QT - 1) / QT;
  fused_conv_kernel<<<nblocks, 128, 0, stream>>>(
      ref_bxyz, ref_feat, query_bxyz, e_ref, e_query, kernel_pos,
      bfrag, out, stats, M, E);

  const int tot = M * C_OUTC;
  bn_kernel<<<(tot + 255) / 256, 256, 0, stream>>>(out, stats, bn_gamma, bn_beta, M);
}
